// LSHAttention_43361989820746
// MI455X (gfx1250) — matrix operand order repaired and checked by reading, not compiled, not run on hardware
//
#include <hip/hip_runtime.h>

// ---------------------------------------------------------------------------
// Types for CDNA5 WMMA (gfx1250, wave32)
// ---------------------------------------------------------------------------
typedef __attribute__((ext_vector_type(16))) __bf16       bf16x16;
typedef __attribute__((ext_vector_type(8)))  float        f32x8;
typedef __attribute__((ext_vector_type(8)))  unsigned int u32x8;

// Problem constants (match reference setup_inputs)
#define BB   2
#define NN   4096
#define DD   1024
#define HH   16
#define GG   4          // Hh hash rounds
#define DK   64
#define SS   64         // CHUNK
#define CC   (NN / SS)  // 64 chunks
#define NBGH (BB * GG * HH)   // 128 independent sort domains
#define NEGV (-1000000000.0f)

static __device__ __forceinline__ unsigned short f2bf(float f) {
    unsigned int u = __float_as_uint(f);
    u = (u + 0x7FFFu + ((u >> 16) & 1u)) >> 16;   // round-to-nearest-even
    return (unsigned short)u;
}

static __device__ __forceinline__ f32x8 wmma_bf16(const u32x8& a, const u32x8& b, f32x8 c) {
    bf16x16 av = __builtin_bit_cast(bf16x16, a);
    bf16x16 bv = __builtin_bit_cast(bf16x16, b);
    return __builtin_amdgcn_wmma_f32_16x16x32_bf16(false, av, false, bv, (short)0, c,
                                                   false, false);
}

// ---------------------------------------------------------------------------
// f32 -> bf16 conversion
// ---------------------------------------------------------------------------
__global__ void cvt_f32_bf16(const float* __restrict__ src,
                             unsigned short* __restrict__ dst, long long n) {
    long long i = (long long)blockIdx.x * 256 + threadIdx.x;
    if (i < n) dst[i] = f2bf(src[i]);
}

// ---------------------------------------------------------------------------
// NT GEMM: C[M,N] = A[M,K] (bf16, row-major) * B[N,K]^T (bf16, row-major) + bias[N]
// Block = 256 threads = 8 waves; block tile 64x64; each wave: 16(M) x 32(N).
// Fragments gathered straight from global — operand matrices are L2-resident
// (192 MB global L2) so re-reads across block tiles are cheap.
// ---------------------------------------------------------------------------
__global__ __launch_bounds__(256)
void gemm_nt_bf16(const unsigned short* __restrict__ A,
                  const unsigned short* __restrict__ Bm,
                  const float* __restrict__ bias,
                  float* __restrict__ C, int M, int N, int K) {
    int tid  = threadIdx.x;
    int wave = tid >> 5, lane = tid & 31;
    int half = lane >> 4, lm = lane & 15;
    int m0 = blockIdx.y * 64 + (wave & 3) * 16;
    int n0 = blockIdx.x * 64 + (wave >> 2) * 32;

    const unsigned short* Arow  = A  + (size_t)(m0 + lm) * K;
    const unsigned short* Brow0 = Bm + (size_t)(n0 + lm) * K;
    const unsigned short* Brow1 = Bm + (size_t)(n0 + 16 + lm) * K;

    f32x8 acc0 = {}, acc1 = {};
    for (int k0 = 0; k0 < K; k0 += 32) {
        u32x8 ap, bp0, bp1;
#pragma unroll
        for (int v = 0; v < 8; ++v) {
            // A layout: lanes 0-15 rows M, K in {0..7,16..23}; lanes 16-31 K+8
            int ka = k0 + ((v & 3) * 2) + (half ? 8 : 0) + ((v & 4) ? 16 : 0);
            ap[v] = *(const unsigned int*)(Arow + ka);
            // B layout: lane = column, K same as A layout
            int kb = k0 + ((v & 3) * 2) + (half ? 8 : 0) + ((v & 4) ? 16 : 0);
            bp0[v] = *(const unsigned int*)(Brow0 + kb);
            bp1[v] = *(const unsigned int*)(Brow1 + kb);
        }
        acc0 = wmma_bf16(ap, bp0, acc0);
        acc1 = wmma_bf16(ap, bp1, acc1);
    }
#pragma unroll
    for (int r = 0; r < 8; ++r) {
        int row = m0 + r + half * 8;        // D: M = r + 8*half, N = lm
        int c0 = n0 + lm, c1 = n0 + 16 + lm;
        C[(size_t)row * N + c0] = acc0[r] + bias[c0];
        C[(size_t)row * N + c1] = acc1[r] + bias[c1];
    }
}

// ---------------------------------------------------------------------------
// invnorm[g,h,m] = 1/||R[g,h,:,m]||
// ---------------------------------------------------------------------------
__global__ void rnorm_kernel(const float* __restrict__ R, float* __restrict__ invn) {
    int i = blockIdx.x * 256 + threadIdx.x;          // < GG*HH*32
    if (i >= GG * HH * 32) return;
    int m = i & 31, gh = i >> 5;
    const float* Rb = R + (size_t)gh * DK * 32 + m;
    float s = 0.f;
    for (int d = 0; d < DK; ++d) { float r = Rb[d * 32]; s += r * r; }
    invn[i] = rsqrtf(s);
}

// ---------------------------------------------------------------------------
// LSH hash + composite sort key:  key = argmax([proj,-proj]) * (N+1) + pos
// One thread per (b,g,h,n).
// ---------------------------------------------------------------------------
__global__ __launch_bounds__(256)
void hash_kernel(const float* __restrict__ qkv, const float* __restrict__ R,
                 const float* __restrict__ invn, unsigned int* __restrict__ keys) {
    int i = blockIdx.x * 256 + threadIdx.x;          // < BB*GG*HH*NN
    int n = i & (NN - 1), h = (i >> 12) & (HH - 1), g = (i >> 16) & (GG - 1), b = i >> 18;
    const float* q  = qkv + ((size_t)(b * NN + n)) * (2 * DD) + h * DK;
    const float* Rb = R + ((size_t)(g * HH + h) * DK) * 32;
    const float* iv = invn + (g * HH + h) * 32;
    float pr[32];
#pragma unroll
    for (int m = 0; m < 32; ++m) pr[m] = 0.f;
    for (int d = 0; d < DK; ++d) {
        float qd = q[d];
        const float* Rr = Rb + d * 32;
#pragma unroll
        for (int m = 0; m < 32; ++m) pr[m] += qd * Rr[m];
    }
#pragma unroll
    for (int m = 0; m < 32; ++m) pr[m] *= iv[m];
    float best = -3.4e38f; int bi = 0;
#pragma unroll
    for (int idx = 0; idx < 64; ++idx) {            // first-occurrence argmax
        float v = (idx < 32) ? pr[idx] : -pr[idx - 32];
        if (v > best) { best = v; bi = idx; }
    }
    keys[(size_t)((b * GG + g) * HH + h) * NN + n] =
        (unsigned)bi * (unsigned)(NN + 1) + (unsigned)n;
}

// ---------------------------------------------------------------------------
// In-LDS bitonic sort of 4096 keys per (b,g,h). Sorting the composite key is
// the argsort: pos = key % 4097, bucket = key / 4097.
// ---------------------------------------------------------------------------
__global__ __launch_bounds__(256)
void sort_kernel(unsigned int* __restrict__ keys) {
    __shared__ unsigned int sk[NN];
    unsigned int* base = keys + (size_t)blockIdx.x * NN;
    for (int i = threadIdx.x; i < NN; i += 256) sk[i] = base[i];
    __syncthreads();
    for (int k = 2; k <= NN; k <<= 1) {
        for (int j = k >> 1; j > 0; j >>= 1) {
#pragma unroll 1
            for (int s = 0; s < NN / 256; ++s) {
                int i = threadIdx.x + s * 256;
                int ixj = i ^ j;
                if (ixj > i) {
                    bool up = (i & k) == 0;
                    unsigned int a = sk[i], b2 = sk[ixj];
                    if ((a > b2) == up) { sk[i] = b2; sk[ixj] = a; }
                }
            }
            __syncthreads();
        }
    }
    for (int i = threadIdx.x; i < NN; i += 256) base[i] = sk[i];
}

// ---------------------------------------------------------------------------
// Chunked LSH attention. One block per (b,g,h,chunk). K rows ARE the sorted
// q rows (shared-QK attention). V is staged TRANSPOSED in LDS so both WMMA
// stages read their fragments as contiguous dword runs (ds_load_b128-able).
// outg[(bgh*N + n)*dk + d] scatter is race-free (perm is a permutation), so
// the whole pipeline is deterministic.
// ---------------------------------------------------------------------------
__global__ __launch_bounds__(256)
void attn_kernel(const float* __restrict__ qkv, const unsigned int* __restrict__ keys,
                 const unsigned char* __restrict__ pad, float* __restrict__ outg) {
    __shared__ unsigned short kb[192][DK];    // bf16 keys(=q) window, row-major 24 KB
    __shared__ unsigned short vbT[DK][192];   // bf16 values window, TRANSPOSED 24 KB
    __shared__ float          sc[SS][192];    // f32 scores                    48 KB
    __shared__ unsigned short att[SS][192];   // bf16 attn probs               24 KB
    __shared__ int bk[192];                   // k-side bucket (-1 = masked)
    __shared__ int bq[SS];                    // q-side raw bucket
    __shared__ int nq[SS];                    // original position of q row
    __shared__ float vq[SS];                  // q-side valid multiplier

    const int c   = blockIdx.x;
    const int bgh = blockIdx.y;
    const int h = bgh & (HH - 1);
    const int b = bgh >> 6;                   // bgh = (b*GG+g)*HH + h
    const int tid = threadIdx.x, wave = tid >> 5, lane = tid & 31;
    const int half = lane >> 4, lm = lane & 15;
    const unsigned int* kset = keys + (size_t)bgh * NN;

    // ---- gather 192-row sorted window, convert to bf16 --------------------
    for (int j = wave; j < 192; j += 8) {
        int p = c * SS - SS + j;
        int bucket = -1;
        float q0 = 0.f, q1 = 0.f, v0 = 0.f, v1 = 0.f;
        if (p >= 0 && p < NN) {
            unsigned int key = kset[p];
            int n  = (int)(key % (unsigned)(NN + 1));
            bucket = (int)(key / (unsigned)(NN + 1));
            const float* row = qkv + ((size_t)(b * NN + n)) * (2 * DD) + h * DK;
            q0 = row[lane * 2];      q1 = row[lane * 2 + 1];
            v0 = row[DD + lane * 2]; v1 = row[DD + lane * 2 + 1];
            if (pad[b * NN + n]) bucket = -1;          // ~valid_both
        }
        kb[j][lane * 2]     = f2bf(q0);
        kb[j][lane * 2 + 1] = f2bf(q1);
        vbT[lane * 2][j]     = f2bf(v0);               // transposed store
        vbT[lane * 2 + 1][j] = f2bf(v1);
        if (lane == 0) {
            bk[j] = bucket;
            if (j >= SS && j < 2 * SS) {               // middle block = q chunk
                int qi = j - SS;
                unsigned int key = kset[p];
                int n = (int)(key % (unsigned)(NN + 1));
                bq[qi] = (int)(key / (unsigned)(NN + 1));
                nq[qi] = n;
                vq[qi] = pad[b * NN + n] ? 0.f : 1.f;
            }
        }
    }
    __syncthreads();

    // ---- scores = q_ch @ k_both^T / sqrt(dk): 4x12 WMMA tiles -------------
    for (int t = wave; t < 48; t += 8) {
        int mi = t & 3, nj = t >> 2;
        f32x8 acc = {};
#pragma unroll
        for (int ks = 0; ks < 2; ++ks) {
            int k0 = ks * 32;
            u32x8 ap, bp;
#pragma unroll
            for (int v = 0; v < 8; ++v) {
                int ka = k0 + ((v & 3) * 2) + (half ? 8 : 0) + ((v & 4) ? 16 : 0);
                ap[v] = *(const unsigned int*)&kb[SS + mi * 16 + lm][ka];
                int kk = k0 + ((v & 3) * 2) + (half ? 8 : 0) + ((v & 4) ? 16 : 0);
                bp[v] = *(const unsigned int*)&kb[nj * 16 + lm][kk];
            }
            acc = wmma_bf16(ap, bp, acc);
        }
#pragma unroll
        for (int r = 0; r < 8; ++r)
            sc[mi * 16 + r + half * 8][nj * 16 + lm] = acc[r] * 0.125f; // 1/sqrt(64)
    }
    __syncthreads();

    // ---- mask + softmax (f32), one thread per q row -----------------------
    if (tid < SS) {
        int qi = tid, myb = bq[qi];
        float mx = -3.4e38f;
        for (int kj = 0; kj < 192; ++kj) {
            float s = sc[qi][kj];
            bool m = (bk[kj] == -1) || (myb != bk[kj]) || (kj == SS + qi);
            s += m ? NEGV : 0.f;                       // scores + mask*NEG
            sc[qi][kj] = s;
            mx = fmaxf(mx, s);
        }
        float sum = 0.f;
        for (int kj = 0; kj < 192; ++kj) {
            float e = __expf(sc[qi][kj] - mx);
            sum += e; sc[qi][kj] = e;
        }
        float inv = 1.f / sum;
        for (int kj = 0; kj < 192; ++kj) att[qi][kj] = f2bf(sc[qi][kj] * inv);
    }
    __syncthreads();

    // ---- out = attn @ v_both: 4x4 WMMA tiles, K = 192 ---------------------
    // B fragment: B[k][d] = vbT[d][k] -> contiguous u32 runs along k.
    for (int t = wave; t < 16; t += 8) {
        int mi = t & 3, nj = t >> 2;
        f32x8 acc = {};
#pragma unroll
        for (int ks = 0; ks < 6; ++ks) {
            int k0 = ks * 32;
            u32x8 ap, bp;
#pragma unroll
            for (int v = 0; v < 8; ++v) {
                int ka = k0 + ((v & 3) * 2) + (half ? 8 : 0) + ((v & 4) ? 16 : 0);
                ap[v] = *(const unsigned int*)&att[mi * 16 + lm][ka];
                int kk = k0 + ((v & 3) * 2) + (half ? 8 : 0) + ((v & 4) ? 16 : 0);
                bp[v] = *(const unsigned int*)&vbT[nj * 16 + lm][kk];
            }
            acc = wmma_bf16(ap, bp, acc);
        }
#pragma unroll
        for (int r = 0; r < 8; ++r) {
            int qi = mi * 16 + r + half * 8;
            float val = acc[r] * vq[qi];               // out *= valid
            outg[((size_t)bgh * NN + nq[qi]) * DK + nj * 16 + lm] = val;
        }
    }
}

// ---------------------------------------------------------------------------
// Mean over hash rounds -> bf16 context, layout (B,N,H*dk) for out projection
// ---------------------------------------------------------------------------
__global__ void reduce_kernel(const float* __restrict__ outg,
                              unsigned short* __restrict__ ctxb) {
    long long i = (long long)blockIdx.x * 256 + threadIdx.x;   // < B*N*D
    if (i >= (long long)BB * NN * DD) return;
    int d = i & (DK - 1);
    int h = (int)(i >> 6) & (HH - 1);
    int n = (int)(i >> 10) & (NN - 1);
    int b = (int)(i >> 22);
    float s = 0.f;
#pragma unroll
    for (int g = 0; g < GG; ++g)
        s += outg[(((size_t)((b * GG + g) * HH + h)) * NN + n) * DK + d];
    ctxb[i] = f2bf(s * 0.25f);
}

// ---------------------------------------------------------------------------
// Host orchestration
// ---------------------------------------------------------------------------
extern "C" void kernel_launch(void* const* d_in, const int* in_sizes, int n_in,
                              void* d_out, int out_size, void* d_ws, size_t ws_size,
                              hipStream_t stream) {
    const float*         x    = (const float*)d_in[0];
    const unsigned char* pad  = (const unsigned char*)d_in[1]; // all-false bools
    const float*         Wqv  = (const float*)d_in[2];
    const float*         bqv  = (const float*)d_in[3];
    const float*         Wout = (const float*)d_in[4];
    const float*         bout = (const float*)d_in[5];
    const float*         R    = (const float*)d_in[6];
    float*               out  = (float*)d_out;

    char* ws = (char*)d_ws;
    size_t off = 0;
    auto alloc = [&](size_t bytes) { size_t r = off; off += (bytes + 255) & ~(size_t)255; return r; };
    unsigned short* xb    = (unsigned short*)(ws + alloc((size_t)BB * NN * DD * 2));
    unsigned short* wqvb  = (unsigned short*)(ws + alloc((size_t)2 * DD * DD * 2));
    unsigned short* woutb = (unsigned short*)(ws + alloc((size_t)DD * DD * 2));
    float*          qkv   = (float*)(ws + alloc((size_t)BB * NN * 2 * DD * 4));
    float*          invn  = (float*)(ws + alloc((size_t)GG * HH * 32 * 4));
    unsigned int*   keys  = (unsigned int*)(ws + alloc((size_t)NBGH * NN * 4));
    float*          outg  = (float*)(ws + alloc((size_t)NBGH * NN * DK * 4));
    unsigned short* ctxb  = (unsigned short*)(ws + alloc((size_t)BB * NN * DD * 2));
    (void)ws_size; (void)n_in; (void)in_sizes; (void)out_size;

    long long nx = (long long)BB * NN * DD;
    cvt_f32_bf16<<<dim3((unsigned)((nx + 255) / 256)), 256, 0, stream>>>(x, xb, nx);
    long long nwqv = (long long)2 * DD * DD;
    cvt_f32_bf16<<<dim3((unsigned)((nwqv + 255) / 256)), 256, 0, stream>>>(Wqv, wqvb, nwqv);
    long long nwo = (long long)DD * DD;
    cvt_f32_bf16<<<dim3((unsigned)((nwo + 255) / 256)), 256, 0, stream>>>(Wout, woutb, nwo);
    rnorm_kernel<<<dim3((GG * HH * 32 + 255) / 256), 256, 0, stream>>>(R, invn);

    // qkv = x @ Wqv^T + bqv : M=8192 N=2048 K=1024
    gemm_nt_bf16<<<dim3(2 * DD / 64, BB * NN / 64), 256, 0, stream>>>(
        xb, wqvb, bqv, qkv, BB * NN, 2 * DD, DD);

    hash_kernel<<<dim3(BB * GG * HH * NN / 256), 256, 0, stream>>>(qkv, R, invn, keys);
    sort_kernel<<<dim3(NBGH), 256, 0, stream>>>(keys);
    attn_kernel<<<dim3(CC, NBGH), 256, 0, stream>>>(qkv, keys, pad, outg);
    reduce_kernel<<<dim3((unsigned)((nx + 255) / 256)), 256, 0, stream>>>(outg, ctxb);

    // out = ctx @ Wout^T + bout : M=8192 N=1024 K=1024
    gemm_nt_bf16<<<dim3(DD / 64, BB * NN / 64), 256, 0, stream>>>(
        ctxb, woutb, bout, out, BB * NN, DD, DD);
}
